// GRUCell_78769700209273
// MI455X (gfx1250) — compile-verified, matrix operand order checked by reading, not run on hardware
//
#include <hip/hip_runtime.h>
#include <hip/hip_bf16.h>

#define B_DIM 8192
#define E_DIM 1024
#define H_DIM 1024

typedef __attribute__((ext_vector_type(16))) __bf16 v16bf;
typedef __attribute__((ext_vector_type(2)))  __bf16 v2bf;
typedef __attribute__((ext_vector_type(8)))  float  v8f;
typedef __attribute__((ext_vector_type(4)))  int    v4i;

#if defined(__HIP_DEVICE_COMPILE__) && __has_builtin(__builtin_amdgcn_global_load_async_to_lds_b128)
#define HAVE_ASYNC_LDS 1
typedef __attribute__((address_space(1))) v4i* gv4i_p;   // global (AS1) v4i*
typedef __attribute__((address_space(3))) v4i* lv4i_p;   // LDS (AS3) v4i*
#else
#define HAVE_ASYNC_LDS 0
#endif

static __device__ __forceinline__ unsigned short f2bf(float f) {
    unsigned int u = __builtin_bit_cast(unsigned int, f);
    u += 0x7FFFu + ((u >> 16) & 1u);   // round-to-nearest-even
    return (unsigned short)(u >> 16);
}

static __device__ __forceinline__ unsigned int pack2(float lo, float hi) {
#if defined(__HIP_DEVICE_COMPILE__) && __has_builtin(__builtin_amdgcn_cvt_pk_bf16_f32)
    v2bf p = __builtin_amdgcn_cvt_pk_bf16_f32(lo, hi);
    return __builtin_bit_cast(unsigned int, p);
#else
    return (unsigned int)f2bf(lo) | ((unsigned int)f2bf(hi) << 16);
#endif
}

// 16-bit WMMA fragment per CDNA5 layout (lanes 0-15: K 0..7 / 16..23 in
// v0..3 / v4..7; lanes 16-31: K 8..15 / 24..31). rowptr = 32-bf16 row for
// this lane's M (A, row-major tile) or N (B, [n][k]-transposed tile).
static __device__ __forceinline__ v16bf load_frag(const unsigned short* rowptr, int half) {
    union { uint4 q[2]; v16bf v; } u;
    u.q[0] = *(const uint4*)(rowptr + half * 8);
    u.q[1] = *(const uint4*)(rowptr + half * 8 + 16);
    return u.v;
}

static __device__ __forceinline__ float sigm(float x) {
    return 1.0f / (1.0f + __expf(-x));
}

#define WMMA_BF16(A, B, C) \
    __builtin_amdgcn_wmma_f32_16x16x32_bf16(false, (A), false, (B), (short)0, (C), false, false)

// 2x2 sub-tile GEMM step: ACC[mi][ni] += Ami @ B(WIDX,ni)
#define GEMM4(ACC, A0, A1, WIDX)                                            \
    do {                                                                    \
        const v16bf b0_ = load_frag(&lW[WIDX][wCol +      mn][0], half);    \
        const v16bf b1_ = load_frag(&lW[WIDX][wCol + 16 + mn][0], half);    \
        ACC[0][0] = WMMA_BF16(A0, b0_, ACC[0][0]);                          \
        ACC[1][0] = WMMA_BF16(A1, b0_, ACC[1][0]);                          \
        ACC[0][1] = WMMA_BF16(A0, b1_, ACC[0][1]);                          \
        ACC[1][1] = WMMA_BF16(A1, b1_, ACC[1][1]);                          \
    } while (0)

// ---------------------------------------------------------------------------
// Phase 1: accU = x@Wxu + h@Whu ; accR = x@Wxr + h@Whr ; accC = x@Wxc
//   u  = sigmoid(accU + bu)      -> ws_u  (f32)
//   hr = h * sigmoid(accR + br)  -> ws_hr (bf16; A-matrix of phase 2)
//   xc = accC                    -> ws_xc (f32)
// Block tile 128(M) x 64(N); 8 waves in 4x2; each wave owns 32x32 (2x2 tiles).
// ---------------------------------------------------------------------------
__global__ __launch_bounds__(256)
void gru_phase1(const float* __restrict__ x, const float* __restrict__ h,
                const float* __restrict__ Wxu, const float* __restrict__ Whu,
                const float* __restrict__ bu,
                const float* __restrict__ Wxr, const float* __restrict__ Whr,
                const float* __restrict__ br,
                const float* __restrict__ Wxc,
                float* __restrict__ u_out, unsigned short* __restrict__ hr_out,
                float* __restrict__ xc_out)
{
    __shared__ alignas(16) unsigned short lX[128][32];     //  8 KB
    __shared__ alignas(16) unsigned short lH[128][32];     //  8 KB
    __shared__ alignas(16) unsigned short lW[5][64][32];   // 20 KB, transposed [n][k]

    const int tid  = threadIdx.x;
    const int lane = tid & 31;
    const int wave = tid >> 5;
    const int rowBase = blockIdx.y * 128;
    const int colBase = blockIdx.x * 64;

    const int wRow = (wave >> 1) * 32;   // 0,32,64,96
    const int wCol = (wave & 1) * 32;    // 0,32
    const int half = lane >> 4;
    const int mn   = lane & 15;

    v8f accU[2][2] = {};
    v8f accR[2][2] = {};
    v8f accC[2][2] = {};

    const float* W[5] = {Wxu, Whu, Wxr, Whr, Wxc};

    for (int kk = 0; kk < E_DIM; kk += 32) {
        __syncthreads();
        // Stage x/h 128x32 f32 tiles -> bf16 LDS. 1024 float4 slots, 4/thread.
        #pragma unroll
        for (int i = 0; i < 4; ++i) {
            const int s = tid + i * 256;
            const int r = s >> 3, cg = s & 7;
            const float4 xv = *(const float4*)(x + (size_t)(rowBase + r) * E_DIM + kk + cg * 4);
            const float4 hv = *(const float4*)(h + (size_t)(rowBase + r) * H_DIM + kk + cg * 4);
            uint2 px; px.x = pack2(xv.x, xv.y); px.y = pack2(xv.z, xv.w);
            uint2 ph; ph.x = pack2(hv.x, hv.y); ph.y = pack2(hv.z, hv.w);
            *(uint2*)&lX[r][cg * 4] = px;
            *(uint2*)&lH[r][cg * 4] = ph;
        }
        // Stage 5 weight 32(k)x64(n) tiles, transposed into [n][k].
        #pragma unroll
        for (int w = 0; w < 5; ++w) {
            #pragma unroll
            for (int i = 0; i < 2; ++i) {
                const int s  = tid + i * 256;      // 512 slots
                const int k  = s >> 4;
                const int n0 = (s & 15) * 4;
                const float4 wv = *(const float4*)(W[w] + (size_t)(kk + k) * H_DIM + colBase + n0);
                lW[w][n0 + 0][k] = f2bf(wv.x);
                lW[w][n0 + 1][k] = f2bf(wv.y);
                lW[w][n0 + 2][k] = f2bf(wv.z);
                lW[w][n0 + 3][k] = f2bf(wv.w);
            }
        }
        __syncthreads();

        const v16bf ax0 = load_frag(&lX[wRow +      mn][0], half);
        const v16bf ax1 = load_frag(&lX[wRow + 16 + mn][0], half);
        const v16bf ah0 = load_frag(&lH[wRow +      mn][0], half);
        const v16bf ah1 = load_frag(&lH[wRow + 16 + mn][0], half);

        GEMM4(accU, ax0, ax1, 0);   // x @ Wxu
        GEMM4(accU, ah0, ah1, 1);   // h @ Whu
        GEMM4(accR, ax0, ax1, 2);   // x @ Wxr
        GEMM4(accR, ah0, ah1, 3);   // h @ Whr
        GEMM4(accC, ax0, ax1, 4);   // x @ Wxc
    }

    // Epilogue: C/D layout -> row = 8*half + j, col = lane&15 (per 16x16 tile).
    #pragma unroll
    for (int ni = 0; ni < 2; ++ni) {
        const int col = colBase + wCol + ni * 16 + mn;
        const float bu_v = bu[col];
        const float br_v = br[col];
        #pragma unroll
        for (int mi = 0; mi < 2; ++mi) {
            #pragma unroll
            for (int j = 0; j < 8; ++j) {
                const int row = rowBase + wRow + mi * 16 + half * 8 + j;
                const size_t idx = (size_t)row * H_DIM + col;
                const float uv = sigm(accU[mi][ni][j] + bu_v);
                const float rv = sigm(accR[mi][ni][j] + br_v);
                const float hv = h[idx];
                u_out[idx]  = uv;
                hr_out[idx] = f2bf(hv * rv);
                xc_out[idx] = accC[mi][ni][j];
            }
        }
    }
}

// ---------------------------------------------------------------------------
// Phase 2: acc = hr @ Whc ; c = u*tanh(xc + acc + bc) + (1-u)*h
// ---------------------------------------------------------------------------
__global__ __launch_bounds__(256)
void gru_phase2(const unsigned short* __restrict__ hr,
                const float* __restrict__ Whc, const float* __restrict__ bc,
                const float* __restrict__ xc, const float* __restrict__ u,
                const float* __restrict__ h, float* __restrict__ out)
{
    __shared__ alignas(16) unsigned short lA[128][32];   // hr tile (bf16)
    __shared__ alignas(16) unsigned short lW[1][64][32]; // Whc tile, [n][k]

    const int tid  = threadIdx.x;
    const int lane = tid & 31;
    const int wave = tid >> 5;
    const int rowBase = blockIdx.y * 128;
    const int colBase = blockIdx.x * 64;

    const int wRow = (wave >> 1) * 32;
    const int wCol = (wave & 1) * 32;
    const int half = lane >> 4;
    const int mn   = lane & 15;

    v8f acc[2][2] = {};

    for (int kk = 0; kk < H_DIM; kk += 32) {
        __syncthreads();
        // Stage hr tile: 128x32 bf16 = 512 x 16B slots, 2/thread.
        #pragma unroll
        for (int i = 0; i < 2; ++i) {
            const int s = tid + i * 256;
            const int r = s >> 2, g = s & 3;
            const unsigned short* src = hr + (size_t)(rowBase + r) * H_DIM + kk + g * 8;
            unsigned short* dst = &lA[r][g * 8];
#if HAVE_ASYNC_LDS
            __builtin_amdgcn_global_load_async_to_lds_b128((gv4i_p)src, (lv4i_p)dst, 0, 0);
#else
            *(uint4*)dst = *(const uint4*)src;
#endif
        }
        // Stage Whc 32(k)x64(n) tile, transposed into [n][k].
        #pragma unroll
        for (int i = 0; i < 2; ++i) {
            const int s  = tid + i * 256;
            const int k  = s >> 4;
            const int n0 = (s & 15) * 4;
            const float4 wv = *(const float4*)(Whc + (size_t)(kk + k) * H_DIM + colBase + n0);
            lW[0][n0 + 0][k] = f2bf(wv.x);
            lW[0][n0 + 1][k] = f2bf(wv.y);
            lW[0][n0 + 2][k] = f2bf(wv.z);
            lW[0][n0 + 3][k] = f2bf(wv.w);
        }
#if HAVE_ASYNC_LDS
#if __has_builtin(__builtin_amdgcn_s_wait_asynccnt)
        __builtin_amdgcn_s_wait_asynccnt(0);
#else
        asm volatile("s_wait_asynccnt 0x0" ::: "memory");
#endif
#endif
        __syncthreads();

        const v16bf a0 = load_frag(&lA[wRow +      mn][0], half);
        const v16bf a1 = load_frag(&lA[wRow + 16 + mn][0], half);
        GEMM4(acc, a0, a1, 0);
    }

    #pragma unroll
    for (int ni = 0; ni < 2; ++ni) {
        const int col = colBase + wCol + ni * 16 + mn;
        const float bc_v = bc[col];
        #pragma unroll
        for (int mi = 0; mi < 2; ++mi) {
            #pragma unroll
            for (int j = 0; j < 8; ++j) {
                const int row = rowBase + wRow + mi * 16 + half * 8 + j;
                const size_t idx = (size_t)row * H_DIM + col;
                const float cp = tanhf(acc[mi][ni][j] + xc[idx] + bc_v);
                const float uv = u[idx];
                const float hv = h[idx];
                out[idx] = uv * cp + (1.0f - uv) * hv;
            }
        }
    }
}

extern "C" void kernel_launch(void* const* d_in, const int* in_sizes, int n_in,
                              void* d_out, int out_size, void* d_ws, size_t ws_size,
                              hipStream_t stream) {
    const float* x   = (const float*)d_in[0];
    const float* h   = (const float*)d_in[1];
    const float* Wxu = (const float*)d_in[2];
    const float* Whu = (const float*)d_in[3];
    const float* bu  = (const float*)d_in[4];
    const float* Wxr = (const float*)d_in[5];
    const float* Whr = (const float*)d_in[6];
    const float* br  = (const float*)d_in[7];
    const float* Wxc = (const float*)d_in[8];
    const float* Whc = (const float*)d_in[9];
    const float* bc  = (const float*)d_in[10];

    float* ws_u  = (float*)d_ws;                                              // 32 MB
    float* ws_xc = ws_u + (size_t)B_DIM * H_DIM;                              // 32 MB
    unsigned short* ws_hr = (unsigned short*)(ws_xc + (size_t)B_DIM * H_DIM); // 16 MB

    dim3 grid(H_DIM / 64, B_DIM / 128);   // 16 x 64 blocks
    gru_phase1<<<grid, 256, 0, stream>>>(x, h, Wxu, Whu, bu, Wxr, Whr, br, Wxc,
                                         ws_u, ws_hr, ws_xc);
    gru_phase2<<<grid, 256, 0, stream>>>(ws_hr, Whc, bc, ws_xc, ws_u, h,
                                         (float*)d_out);
}
